// Summarizer_84250078478346
// MI455X (gfx1250) — matrix operand order repaired and checked by reading, not compiled, not run on hardware
//
#include <hip/hip_runtime.h>

// ---------------------------------------------------------------------------
// Summarizer for MI455X (gfx1250): fused conv|.|pool front-end + GEMM tower,
// all matmuls on v_wmma_f32_16x16x32_f16 (wave32 WMMA).
// LDS B-matrices stored t-major so every WMMA B fragment is 2x ds_load_b128.
// ---------------------------------------------------------------------------

typedef __attribute__((ext_vector_type(16))) _Float16 v16h;
typedef __attribute__((ext_vector_type(8)))  _Float16 v8h;
typedef __attribute__((ext_vector_type(8)))  float    v8f;

#define MODEL_DIM 128
#define N_SAMPLES 32768
#define KERNEL_SIZE 256
#define N_FRAMES 128
#define BATCH 32

// workspace byte offsets (256B aligned)
#define OFF_FB    0u          // fb16:    128*256 f16            = 65536 B
#define OFF_RW    65536u      // rw16:    128*192 f16 (padded)   = 49152 B
#define OFF_DA    114688u     // dilA16:  5*128*384 f16          = 491520 B
#define OFF_PW    606208u     // pw16:    5*128*128 f16          = 163840 B
#define OFF_PART  770048u     // partial: 32*128*128 f32         = 2097152 B
#define OFF_HA    2867200u    // hA:      32*128*128 f32         = 2097152 B
#define OFF_HB    4964352u    // hB:      32*128*128 f32         = 2097152 B

static __device__ inline v8f wmma_f16(v16h a, v16h b, v8f c) {
  // v_wmma_f32_16x16x32_f16  (neg_a, A, neg_b, B, c_mod, C, reuse_a, reuse_b)
  return __builtin_amdgcn_wmma_f32_16x16x32_f16(false, a, false, b, (short)0, c,
                                                false, false);
}

// A-matrix fragment (16x32 f16): lane m = lane&15,
// khalf = lane>>4. v0-7 hold K = khalf*16 + 0..15 (contiguous, matches B).
// `col` = kstep*32 + khalf*16 ; rows are contiguous in memory -> two b128 loads.
static __device__ inline v16h load_a_frag(const _Float16* row, int col) {
  v8h lo = *(const v8h*)(row + col);
  v8h hi = *(const v8h*)(row + col + 8);
  return __builtin_shufflevector(lo, hi, 0, 1, 2, 3, 4, 5, 6, 7,
                                 8, 9, 10, 11, 12, 13, 14, 15);
}

// B-matrix fragment (32x16 f16): lane n = lane&15, khalf = lane>>4,
// VGPR j holds K = khalf*16 + 2j, 2j+1 -> 16 K-consecutive halves per lane.
// Unknown alignment (k_frontend sliding window): scalar fill, compiler merges.
static __device__ inline v16h load_b_contig(const _Float16* p) {
  union { v16h v; _Float16 h[16]; } u;
#pragma unroll
  for (int q = 0; q < 16; ++q) u.h[q] = p[q];
  return u.v;
}

// 16B-aligned contiguous B fragment: exactly two ds_load_b128.
static __device__ inline v16h load_b_aligned(const _Float16* p) {
  v8h lo = *(const v8h*)(p);
  v8h hi = *(const v8h*)(p + 8);
  return __builtin_shufflevector(lo, hi, 0, 1, 2, 3, 4, 5, 6, 7,
                                 8, 9, 10, 11, 12, 13, 14, 15);
}

// ---------------------------------------------------------------------------
// Kernel 0: weight conversion / re-layout to f16
// ---------------------------------------------------------------------------
__global__ __launch_bounds__(256) void k_prep(
    const float* __restrict__ fb, const float* __restrict__ rw,
    const float* __restrict__ dw, const float* __restrict__ pw,
    _Float16* __restrict__ fb16, _Float16* __restrict__ rw16,
    _Float16* __restrict__ dilA16, _Float16* __restrict__ pw16) {
  const int total = 32768 + 24576 + 245760 + 81920;
  for (int i = blockIdx.x * blockDim.x + threadIdx.x; i < total;
       i += gridDim.x * blockDim.x) {
    if (i < 32768) {
      fb16[i] = (_Float16)fb[i];
    } else {
      int i2 = i - 32768;
      if (i2 < 24576) {  // reduce_w (128 x 161) -> (128 x 192) zero-padded
        int o = i2 / 192, k = i2 % 192;
        rw16[i2] = (k < 161) ? (_Float16)rw[o * 161 + k] : (_Float16)0.f;
      } else {
        int i3 = i2 - 24576;
        if (i3 < 245760) {  // dil_w (5,128,128,3) -> A1[s][o][j*128+c]
          int kk = i3 % 384;
          int r = (i3 / 384) % 128;
          int s5 = i3 / (384 * 128);
          int j = kk >> 7, c = kk & 127;
          dilA16[i3] = (_Float16)dw[(((size_t)s5 * 128 + r) * 128 + c) * 3 + j];
        } else {  // pw_w (5,128,128) straight copy
          int i4 = i3 - 245760;
          pw16[i4] = (_Float16)pw[i4];
        }
      }
    }
  }
}

// ---------------------------------------------------------------------------
// Kernel 1: fused conv + |.| + partial pooling.
// Block = (chunk c, batch b). Computes partial[b][c][ch] =
//   sum_{s in [c*256, c*256+256)} | sum_k fb[ch,k] * x[s+k-128] |
// via WMMA GEMM: M=128 (ch), N=256 (pos), K=256 (taps).
// ---------------------------------------------------------------------------
__global__ __launch_bounds__(256) void k_frontend(
    const float* __restrict__ x, const _Float16* __restrict__ fb16,
    float* __restrict__ partial) {
  const int chunk = blockIdx.x;
  const int b = blockIdx.y;
  __shared__ alignas(16) _Float16 xl[512];
  __shared__ float psum[128];
  const int tid = threadIdx.x;

  if (tid < 128) psum[tid] = 0.f;
#pragma unroll
  for (int i = tid; i < 512; i += 256) {
    int s = chunk * 256 - 128 + i;  // x index with conv padding of 128
    float v = (s >= 0 && s < N_SAMPLES) ? x[(size_t)b * N_SAMPLES + s] : 0.f;
    xl[i] = (_Float16)v;
  }
  __syncthreads();

  const int lane = tid & 31;
  const int wave = tid >> 5;
  const int khalf = lane >> 4;
  const int ncol = lane & 15;
  const int mtile = wave;  // 8 waves -> 8 M-tiles of 16 channels

  const _Float16* arow = fb16 + (mtile * 16 + ncol) * KERNEL_SIZE;
  v16h Af[8];
#pragma unroll
  for (int ks = 0; ks < 8; ++ks) Af[ks] = load_a_frag(arow, ks * 32 + khalf * 16);

  float rowsum[8] = {0.f, 0.f, 0.f, 0.f, 0.f, 0.f, 0.f, 0.f};
  for (int nt = 0; nt < 16; ++nt) {
    v8f acc = {0.f, 0.f, 0.f, 0.f, 0.f, 0.f, 0.f, 0.f};
#pragma unroll
    for (int ks = 0; ks < 8; ++ks) {
      // B[k][n] = xl[n + k] : contiguous 16 halves per fragment
      int base = nt * 16 + ncol + ks * 32 + khalf * 16;
      v16h bf = load_b_contig(&xl[base]);
      acc = wmma_f16(Af[ks], bf, acc);
    }
#pragma unroll
    for (int v = 0; v < 8; ++v) rowsum[v] += fabsf(acc[v]);
  }
#pragma unroll
  for (int v = 0; v < 8; ++v)
    atomicAdd(&psum[mtile * 16 + v + 8 * khalf], rowsum[v]);
  __syncthreads();
  if (tid < 128)
    partial[((size_t)b * 128 + chunk) * 128 + tid] = psum[tid];
}

// ---------------------------------------------------------------------------
// Kernel 2: pooled + pos-encoding concat + reduce GEMM (K=161 padded to 192).
// LDS B stored t-major: Bm[t][192].
// ---------------------------------------------------------------------------
__global__ __launch_bounds__(256) void k_reduce(
    const float* __restrict__ partial, const _Float16* __restrict__ rw16,
    const float* __restrict__ reduce_b, float* __restrict__ h0) {
  const int b = blockIdx.x;
  __shared__ alignas(16) _Float16 Bm[128 * 192];
  const int tid = threadIdx.x;
  const float PI = 3.14159265358979323846f;

  for (int i = tid; i < 128 * 192; i += 256) {
    int t = i / 192, k = i - t * 192;
    float v;
    if (k < 128) {  // pooled: avg of chunk t-1 and chunk t partials / 512
      float p1 = partial[((size_t)b * 128 + t) * 128 + k];
      float p0 = (t > 0) ? partial[((size_t)b * 128 + (t - 1)) * 128 + k] : 0.f;
      v = (p0 + p1) * (1.0f / 512.0f);
    } else if (k < 161) {  // positional encoding
      int c = k - 128;
      float tl = -PI + (float)t * (2.0f * PI / 127.0f);
      if (c == 0) {
        v = tl;
      } else {
        int fi = (c - 1) >> 1;
        float a = (float)(1 << fi) * tl;
        v = (((c - 1) & 1) == 0) ? sinf(a) : cosf(a);
      }
    } else {
      v = 0.f;  // K padding
    }
    Bm[i] = (_Float16)v;
  }
  __syncthreads();

  const int lane = tid & 31, wave = tid >> 5;
  const int khalf = lane >> 4, ncol = lane & 15;
  const int mtile = wave;
  const _Float16* arow = rw16 + (mtile * 16 + ncol) * 192;
  const _Float16* brow = Bm;  // [t][192]

  for (int nt = 0; nt < 8; ++nt) {
    const int t = nt * 16 + ncol;
    v8f acc = {0.f, 0.f, 0.f, 0.f, 0.f, 0.f, 0.f, 0.f};
#pragma unroll
    for (int ks = 0; ks < 6; ++ks) {
      v16h a = load_a_frag(arow, ks * 32 + khalf * 16);
      v16h bf = load_b_aligned(&brow[t * 192 + ks * 32 + khalf * 16]);
      acc = wmma_f16(a, bf, acc);
    }
#pragma unroll
    for (int v = 0; v < 8; ++v) {
      int o = mtile * 16 + v + 8 * khalf;
      h0[((size_t)b * 128 + o) * 128 + t] = acc[v] + reduce_b[o];
    }
  }
}

// ---------------------------------------------------------------------------
// Kernel 3: one dilated residual stage.
//   y = conv3(h, dilation d) + b1       (3 accumulated K=128 GEMM passes)
//   z = pw @ y + b2 ;  h' = leaky_relu(z + h, 0.2)
// LDS matrices stored t-major: Bl[t][128] (k = channel c), Yl[t][128] (k = o).
// ---------------------------------------------------------------------------
__global__ __launch_bounds__(256) void k_stage(
    const float* __restrict__ h_in, float* __restrict__ h_out,
    const _Float16* __restrict__ A1,  // [128][384]  (o, j*128+c)
    const _Float16* __restrict__ A2,  // [128][128]
    const float* __restrict__ b1, const float* __restrict__ b2, int d) {
  const int b = blockIdx.x;
  __shared__ alignas(16) _Float16 Bl[128 * 128];
  __shared__ alignas(16) _Float16 Yl[128 * 128];
  const int tid = threadIdx.x;
  const int lane = tid & 31, wave = tid >> 5;
  const int khalf = lane >> 4, ncol = lane & 15;
  const int mtile = wave;
  const float* hb = h_in + (size_t)b * 128 * 128;

  v8f acc[8];
#pragma unroll
  for (int nt = 0; nt < 8; ++nt)
#pragma unroll
    for (int v = 0; v < 8; ++v) acc[nt][v] = 0.f;

  for (int j = 0; j < 3; ++j) {
    __syncthreads();
    for (int i = tid; i < 128 * 128; i += 256) {
      int t = i >> 7, c = i & 127;  // Bl[t][c] = h[c][t + (j-1)d]
      int tt = t + (j - 1) * d;
      float v = (tt >= 0 && tt < 128) ? hb[c * 128 + tt] : 0.f;
      Bl[i] = (_Float16)v;
    }
    __syncthreads();
    const _Float16* arow = A1 + (mtile * 16 + ncol) * 384 + j * 128;
    for (int nt = 0; nt < 8; ++nt) {
      const int t = nt * 16 + ncol;
#pragma unroll
      for (int ks = 0; ks < 4; ++ks) {
        v16h a = load_a_frag(arow, ks * 32 + khalf * 16);
        v16h bf = load_b_aligned(&Bl[t * 128 + ks * 32 + khalf * 16]);
        acc[nt] = wmma_f16(a, bf, acc[nt]);
      }
    }
  }
  __syncthreads();
#pragma unroll
  for (int nt = 0; nt < 8; ++nt)
#pragma unroll
    for (int v = 0; v < 8; ++v) {
      int o = mtile * 16 + v + 8 * khalf;
      int t = nt * 16 + ncol;
      Yl[t * 128 + o] = (_Float16)(acc[nt][v] + b1[o]);  // t-major
    }
  __syncthreads();

  const _Float16* arow2 = A2 + (mtile * 16 + ncol) * 128;
  for (int nt = 0; nt < 8; ++nt) {
    const int t = nt * 16 + ncol;
    v8f a2 = {0.f, 0.f, 0.f, 0.f, 0.f, 0.f, 0.f, 0.f};
#pragma unroll
    for (int ks = 0; ks < 4; ++ks) {
      v16h a = load_a_frag(arow2, ks * 32 + khalf * 16);
      v16h bf = load_b_aligned(&Yl[t * 128 + ks * 32 + khalf * 16]);
      a2 = wmma_f16(a, bf, a2);
    }
#pragma unroll
    for (int v = 0; v < 8; ++v) {
      int o = mtile * 16 + v + 8 * khalf;
      float r = a2[v] + b2[o] + hb[o * 128 + t];
      h_out[(size_t)b * 128 * 128 + o * 128 + t] = (r > 0.f) ? r : 0.2f * r;
    }
  }
}

// ---------------------------------------------------------------------------
// Kernel 4: attention logits, softmax over channels, rank-based top-16,
// gather (channel index reused as time index, per reference), write outputs.
// ---------------------------------------------------------------------------
__global__ __launch_bounds__(128) void k_attend(
    const float* __restrict__ h, const float* __restrict__ aw,
    const float* __restrict__ ab, float* __restrict__ out) {
  const int b = blockIdx.x;
  const int ch = threadIdx.x;
  __shared__ float sl[128], se[128];
  const float* hb = h + (size_t)b * 128 * 128;

  float lg = ab[0];
  for (int t = 0; t < 128; ++t) lg += hb[ch * 128 + t] * aw[t];
  sl[ch] = lg;
  __syncthreads();
  float m = -1e30f;
  for (int jx = 0; jx < 128; ++jx) m = fmaxf(m, sl[jx]);
  float e = expf(lg - m);
  se[ch] = e;
  __syncthreads();
  float s = 0.f;
  for (int jx = 0; jx < 128; ++jx) s += se[jx];
  float attn = e / s;
  __syncthreads();
  sl[ch] = attn;
  __syncthreads();
  int rank = 0;
  for (int jx = 0; jx < 128; ++jx) {
    float aj = sl[jx];
    rank += (aj > attn) || (aj == attn && jx < ch);
  }
  if (rank < 16) {
    // gathered[b, rank, c] = h[b, c, ch] * attn
    for (int c = 0; c < 128; ++c)
      out[(size_t)b * 2048 + rank * 128 + c] = hb[c * 128 + ch] * attn;
    out[(size_t)BATCH * 2048 + b * 16 + rank] = (float)ch;  // idx output
  }
}

// ---------------------------------------------------------------------------
extern "C" void kernel_launch(void* const* d_in, const int* in_sizes, int n_in,
                              void* d_out, int out_size, void* d_ws,
                              size_t ws_size, hipStream_t stream) {
  const float* x        = (const float*)d_in[0];
  const float* fb       = (const float*)d_in[1];
  const float* reduce_w = (const float*)d_in[2];
  const float* reduce_b = (const float*)d_in[3];
  const float* dil_w    = (const float*)d_in[4];
  const float* dil_b    = (const float*)d_in[5];
  const float* pw_w     = (const float*)d_in[6];
  const float* pw_b     = (const float*)d_in[7];
  const float* attend_w = (const float*)d_in[8];
  const float* attend_b = (const float*)d_in[9];
  (void)in_sizes; (void)n_in; (void)out_size; (void)ws_size;

  char* ws = (char*)d_ws;
  _Float16* fb16   = (_Float16*)(ws + OFF_FB);
  _Float16* rw16   = (_Float16*)(ws + OFF_RW);
  _Float16* dilA16 = (_Float16*)(ws + OFF_DA);
  _Float16* pw16   = (_Float16*)(ws + OFF_PW);
  float* partial   = (float*)(ws + OFF_PART);
  float* hA        = (float*)(ws + OFF_HA);
  float* hB        = (float*)(ws + OFF_HB);

  k_prep<<<dim3(1504), dim3(256), 0, stream>>>(fb, reduce_w, dil_w, pw_w, fb16,
                                               rw16, dilA16, pw16);

  k_frontend<<<dim3(128, BATCH), dim3(256), 0, stream>>>(x, fb16, partial);

  k_reduce<<<dim3(BATCH), dim3(256), 0, stream>>>(partial, rw16, reduce_b, hA);

  const int dil[5] = {1, 3, 9, 27, 1};
  float* cur = hA;
  float* nxt = hB;
  for (int i = 0; i < 5; ++i) {
    k_stage<<<dim3(BATCH), dim3(256), 0, stream>>>(
        cur, nxt, dilA16 + (size_t)i * 128 * 384, pw16 + (size_t)i * 128 * 128,
        dil_b + i * 128, pw_b + i * 128, dil[i]);
    float* tmp = cur; cur = nxt; nxt = tmp;
  }
  // after 5 stages, result is in `cur` (hB)
  k_attend<<<dim3(BATCH), dim3(128), 0, stream>>>(cur, attend_w, attend_b,
                                                  (float*)d_out);
}
